// VectorQuantizer_6657199309296
// MI455X (gfx1250) — matrix operand order repaired and checked by reading, not compiled, not run on hardware
//
#include <hip/hip_runtime.h>
#include <hip/hip_bf16.h>

// ---- problem constants (from reference) ----
#define N_E      1024
#define E_DIM    256
#define HW       4096          // 64*64
#define NB       16            // batch
#define NTOT     65536         // NB*HW rows of zf
#define OUT_ELEMS 16777216     // 16*256*64*64
#define LOSS_OFF  OUT_ELEMS
#define IDX_OFF   (OUT_ELEMS + 1)

typedef __bf16 v16bf __attribute__((ext_vector_type(16)));
typedef float  v8f   __attribute__((ext_vector_type(8)));

union AB32 { uint4 u4[2]; v16bf v; };

__device__ __forceinline__ unsigned short f32_to_bf16_rne(float f) {
    unsigned int u = __float_as_uint(f);
    u = (u + 0x7FFFu + ((u >> 16) & 1u)) >> 16;
    return (unsigned short)u;
}

// ---------------------------------------------------------------------------
// Kernel 1: codebook fp32 -> bf16 (RNE) + per-code squared norms of the
// *rounded* values (consistent with the bf16 dot products used for argmin).
// ---------------------------------------------------------------------------
__global__ void vq_prep(const float* __restrict__ cb,
                        unsigned short* __restrict__ cbBf,
                        float* __restrict__ norms) {
    const int code = blockIdx.x;
    const int t    = threadIdx.x;           // 0..255 == channel
    float v = cb[code * E_DIM + t];
    unsigned short h = f32_to_bf16_rne(v);
    cbBf[code * E_DIM + t] = h;
    float vr = __uint_as_float(((unsigned int)h) << 16);   // rounded value
    __shared__ float s[256];
    s[t] = vr * vr;
    __syncthreads();
    #pragma unroll
    for (int st = 128; st > 0; st >>= 1) {
        if (t < st) s[t] += s[t + st];
        __syncthreads();
    }
    if (t == 0) norms[code] = s[0];
}

// ---------------------------------------------------------------------------
// Kernel 2: main VQ. One workgroup = 16 consecutive zf rows x all 1024 codes.
// 8 waves; wave w owns codes [w*128, w*128+128). 64 WMMAs per wave.
// ---------------------------------------------------------------------------
__global__ void __launch_bounds__(256, 2)
vq_main(const float* __restrict__ z,
        const float* __restrict__ cb,            // fp32 codebook (gather/loss)
        const unsigned short* __restrict__ cbBf, // bf16 codebook (WMMA B)
        const float* __restrict__ norms,
        float* __restrict__ outF,
        float* __restrict__ partials) {
    // LDS tiles
    __shared__ __align__(16) unsigned short zTile[16 * 264]; // bf16, padded row
    __shared__ __align__(16) float zF32[16][260];            // fp32 copy
    __shared__ float redS[8][16];
    __shared__ int   redI[8][16];
    __shared__ int   finalIdx[16];
    __shared__ float rbuf[256];

    const int t     = threadIdx.x;
    const int lane  = t & 31;
    const int wave  = t >> 5;
    const int m     = lane & 15;
    const int half  = lane >> 4;

    const int nbase = blockIdx.x * 16;        // first zf row of this tile
    const int b     = nbase >> 12;            // / HW
    const int sbase = nbase & (HW - 1);

    // ---- Phase 1: stage z tile (16 rows x 256 ch) into LDS, coalesced on n
    {
        const int row = t & 15;
        const int c0  = t >> 4;
        #pragma unroll
        for (int it = 0; it < 16; ++it) {
            const int c = it * 16 + c0;
            const float v = z[(((size_t)b * E_DIM + c) * HW) + sbase + row];
            zF32[row][c] = v;
            zTile[row * 264 + c] = f32_to_bf16_rne(v);
        }
    }
    __syncthreads();

    // ---- Phase 2: WMMA over this wave's 8 code tiles
    float minS[8];
    int   minI[8];
    #pragma unroll
    for (int r = 0; r < 8; ++r) { minS[r] = 3.4e38f; minI[r] = 0; }

    const int ct0 = wave * 8;
    for (int ct = ct0; ct < ct0 + 8; ++ct) {
        const int code = ct * 16 + m;                  // this lane's B column
        const unsigned short* bp = cbBf + (size_t)code * E_DIM;
        v8f acc = {0.f, 0.f, 0.f, 0.f, 0.f, 0.f, 0.f, 0.f};
        #pragma unroll
        for (int kb = 0; kb < E_DIM; kb += 32) {
            // A fragment (16x32 bf16): lane<16 -> K {kb..kb+7, kb+16..kb+23}
            //                          lane>=16 -> K {kb+8..kb+15, kb+24..kb+31}
            AB32 a, bb;
            a.u4[0] = *(const uint4*)&zTile[m * 264 + kb + half * 8];
            a.u4[1] = *(const uint4*)&zTile[m * 264 + kb + 16 + half * 8];
            // B fragment (32x16 bf16): lane holds column N=m, K {kb+half*8..+7, kb+16+half*8..+7}
            bb.u4[0] = *(const uint4*)&bp[kb + half * 8];
            bb.u4[1] = *(const uint4*)&bp[kb + 16 + half * 8];
            acc = __builtin_amdgcn_wmma_f32_16x16x32_bf16(
                      false, a.v, false, bb.v, (short)0, acc, false, false);
        }
        const float cn = norms[code];
        #pragma unroll
        for (int r = 0; r < 8; ++r) {     // C layout: VGPR r -> row half*8+r, col m
            const float s = cn - 2.0f * acc[r];
            if (s < minS[r] || (s == minS[r] && code < minI[r])) {
                minS[r] = s; minI[r] = code;
            }
        }
    }

    // cross-lane (code-column) argmin within each 16-lane half
    #pragma unroll
    for (int off = 1; off < 16; off <<= 1) {
        #pragma unroll
        for (int r = 0; r < 8; ++r) {
            const float oS = __shfl_xor(minS[r], off, 32);
            const int   oI = __shfl_xor(minI[r], off, 32);
            if (oS < minS[r] || (oS == minS[r] && oI < minI[r])) {
                minS[r] = oS; minI[r] = oI;
            }
        }
    }
    if (m == 0) {                              // lane 0 -> rows 0..7, lane 16 -> 8..15
        #pragma unroll
        for (int r = 0; r < 8; ++r) {
            redS[wave][half * 8 + r] = minS[r];
            redI[wave][half * 8 + r] = minI[r];
        }
    }
    __syncthreads();

    // ---- Phase 3: reduce across the 8 waves, emit indices
    if (t < 16) {
        float bs = redS[0][t];
        int   bi = redI[0][t];
        #pragma unroll
        for (int w = 1; w < 8; ++w) {
            const float s = redS[w][t];
            const int   i = redI[w][t];
            if (s < bs || (s == bs && i < bi)) { bs = s; bi = i; }
        }
        finalIdx[t] = bi;
        outF[IDX_OFF + nbase + t] = (float)bi;  // indices (float-typed output)
    }
    __syncthreads();

    // ---- Phase 4: gather fp32 codebook rows, write z_q (== straight-through
    // output), accumulate exact fp32 squared error
    float part = 0.f;
    {
        const int row = t & 15;
        const int c0  = t >> 4;
        const int code = finalIdx[row];
        const float* qrow = cb + (size_t)code * E_DIM;
        #pragma unroll
        for (int it = 0; it < 16; ++it) {
            const int c = it * 16 + c0;
            const float q = qrow[c];
            const float d = q - zF32[row][c];
            part += d * d;
            outF[(((size_t)b * E_DIM + c) * HW) + sbase + row] = q;
        }
    }
    rbuf[t] = part;
    __syncthreads();
    #pragma unroll
    for (int st = 128; st > 0; st >>= 1) {
        if (t < st) rbuf[t] += rbuf[t + st];
        __syncthreads();
    }
    if (t == 0) partials[blockIdx.x] = rbuf[0];
}

// ---------------------------------------------------------------------------
// Kernel 3: deterministic fixed-order reduction of 4096 partials -> loss
// codebook_loss = q_latent + BETA*e_latent = (1+0.25) * MSE (values identical)
// ---------------------------------------------------------------------------
__global__ void vq_finish(const float* __restrict__ partials,
                          float* __restrict__ lossOut) {
    __shared__ float s[256];
    const int t = threadIdx.x;
    float acc = 0.f;
    #pragma unroll
    for (int i = 0; i < 16; ++i) acc += partials[t + i * 256];
    s[t] = acc;
    __syncthreads();
    #pragma unroll
    for (int st = 128; st > 0; st >>= 1) {
        if (t < st) s[t] += s[t + st];
        __syncthreads();
    }
    if (t == 0) lossOut[0] = s[0] * 1.25f / (float)OUT_ELEMS;
}

extern "C" void kernel_launch(void* const* d_in, const int* in_sizes, int n_in,
                              void* d_out, int out_size, void* d_ws, size_t ws_size,
                              hipStream_t stream) {
    const float* z  = (const float*)d_in[0];   // [16,256,64,64] fp32
    const float* cb = (const float*)d_in[1];   // [1024,256] fp32
    float* outF = (float*)d_out;

    // workspace layout: bf16 codebook | code norms | per-WG partial SSE
    unsigned short* cbBf = (unsigned short*)d_ws;                       // 512 KB
    float* norms    = (float*)((char*)d_ws + (size_t)N_E * E_DIM * 2);  // 4 KB
    float* partials = norms + N_E;                                      // 16 KB

    vq_prep  <<<N_E,        256, 0, stream>>>(cb, cbBf, norms);
    vq_main  <<<NTOT / 16,  256, 0, stream>>>(z, cb, cbBf, norms, outF, partials);
    vq_finish<<<1,          256, 0, stream>>>(partials, outF + LOSS_OFF);
}
